// CAM_Factorized_Module_6365141532877
// MI455X (gfx1250) — compile-verified, matrix operand order checked by reading, not run on hardware
//
#include <hip/hip_runtime.h>

// ---- CDNA5 (gfx1250) wave32 WMMA types ----
typedef __bf16 bf16_t;
typedef __attribute__((ext_vector_type(8)))  __bf16 v8bf;
typedef __attribute__((ext_vector_type(16))) __bf16 v16bf;
typedef __attribute__((ext_vector_type(8)))  float  v8f;

#define WMMA_BF16(A, Bm, Cm) \
  __builtin_amdgcn_wmma_f32_16x16x32_bf16(false, (A), false, (Bm), (short)0, (Cm), false, false)

// Problem constants (reference: B=8, C=256, P*H*W=12288, heads=8, Ch=32)
#define B_      8
#define C_      256
#define N_      12288
#define H_      8
#define QKV_LD  768   // 3*C

// ---- bf16 helpers (RNE) ----
__device__ __forceinline__ unsigned short f2bfraw(float f) {
  unsigned u = __builtin_bit_cast(unsigned, f);
  unsigned r = u + 0x7FFFu + ((u >> 16) & 1u);
  return (unsigned short)(r >> 16);
}
__device__ __forceinline__ float bfr2f(unsigned short h) {
  unsigned u = ((unsigned)h) << 16;
  return __builtin_bit_cast(float, u);
}
__device__ __forceinline__ float bfe2f(bf16_t b) {
  return bfr2f(__builtin_bit_cast(unsigned short, b));
}

// Concatenate two contiguous 8x bf16 (16B) chunks into a 16x bf16 fragment.
// Per-lane fragment K layout (ISA 7.12.2, 16-bit 16x16x32):
//   chunk0 = K [8*(lane>>4), +7],  chunk1 = K [16 + 8*(lane>>4), +7]
__device__ __forceinline__ v16bf frag_concat(v8bf lo, v8bf hi) {
  return __builtin_shufflevector(lo, hi, 0,1,2,3,4,5,6,7,8,9,10,11,12,13,14,15);
}
__device__ __forceinline__ v8bf ld8(const unsigned short* p) {
  return *(const v8bf*)p;   // 16B vector load (global or LDS)
}

// ===========================================================================
// Kernel 0: weight prep — dst[col*K + k] = bf16(src[k*cols + col])
// ===========================================================================
__global__ __launch_bounds__(256)
void transpose_bf16_kernel(const float* __restrict__ src, unsigned short* __restrict__ dst,
                           int K, int cols) {
  const int idx = blockIdx.x * 256 + threadIdx.x;
  const int k = idx % K, col = idx / K;
  dst[idx] = f2bfraw(src[(size_t)k * cols + col]);
}

// ===========================================================================
// Kernel 1: qkv[b*N+n, 0:768] = x1 @ W_qkv + b_qkv   (x1[n,c] = x[b,c,n])
// grid (768, 12), block 256 (8 waves). Wave = 16 rows x 64 cols, K=256.
// ===========================================================================
__global__ __launch_bounds__(256)
void qkv_gemm_kernel(const float* __restrict__ x, const unsigned short* __restrict__ WqkvT,
                     const float* __restrict__ bqkv, unsigned short* __restrict__ qkv) {
  __shared__ __align__(16) unsigned short aLds[128][40];   // 32 K x 128 n, +8 pad

  const int lane = threadIdx.x & 31;
  const int wave = threadIdx.x >> 5;
  const int gmBase = blockIdx.x * 128;
  const int rowBase = gmBase + wave * 16;
  const int col0 = blockIdx.y * 64;
  const int b  = gmBase / N_;
  const int n0 = gmBase - b * N_;
  const float* xb = x + (size_t)b * C_ * N_ + n0;

  const int m  = lane & 15;
  const int hi = lane >> 4;
  const unsigned short* wcol = WqkvT + (size_t)(col0 + m) * C_;

  v8f acc0 = {}, acc1 = {}, acc2 = {}, acc3 = {};

  for (int k0 = 0; k0 < C_; k0 += 32) {
#pragma unroll
    for (int i = 0; i < 4; ++i) {
      const int vecIdx = threadIdx.x + i * 256;       // 0..1023 float4s
      const int kk = vecIdx >> 5;                     // 0..31
      const int jj = (vecIdx & 31) * 4;               // 0..124
      const float4 f = *(const float4*)(xb + (size_t)(k0 + kk) * N_ + jj);
      aLds[jj + 0][kk] = f2bfraw(f.x);
      aLds[jj + 1][kk] = f2bfraw(f.y);
      aLds[jj + 2][kk] = f2bfraw(f.z);
      aLds[jj + 3][kk] = f2bfraw(f.w);
    }
    __syncthreads();

    const unsigned short* ar = &aLds[wave * 16 + m][hi * 8];
    const v16bf a = frag_concat(ld8(ar), ld8(ar + 16));

    const unsigned short* wk = wcol + k0 + hi * 8;
    const v16bf b0 = frag_concat(ld8(wk),           ld8(wk + 16));
    const v16bf b1 = frag_concat(ld8(wk + 16 * C_), ld8(wk + 16 * C_ + 16));
    const v16bf b2 = frag_concat(ld8(wk + 32 * C_), ld8(wk + 32 * C_ + 16));
    const v16bf b3 = frag_concat(ld8(wk + 48 * C_), ld8(wk + 48 * C_ + 16));

    acc0 = WMMA_BF16(a, b0, acc0);
    acc1 = WMMA_BF16(a, b1, acc1);
    acc2 = WMMA_BF16(a, b2, acc2);
    acc3 = WMMA_BF16(a, b3, acc3);
    __syncthreads();
  }

  const int mhi = hi << 3;
  const float bb0 = bqkv[col0 + m];
  const float bb1 = bqkv[col0 + 16 + m];
  const float bb2 = bqkv[col0 + 32 + m];
  const float bb3 = bqkv[col0 + 48 + m];
#pragma unroll
  for (int r = 0; r < 8; ++r) {
    const size_t rowOff = (size_t)(rowBase + mhi + r) * QKV_LD + col0 + m;
    qkv[rowOff]      = f2bfraw(acc0[r] + bb0);
    qkv[rowOff + 16] = f2bfraw(acc1[r] + bb1);
    qkv[rowOff + 32] = f2bfraw(acc2[r] + bb2);
    qkv[rowOff + 48] = f2bfraw(acc3[r] + bb3);
  }
}

// ===========================================================================
// Kernel 2: per-(b,h,ch) column online softmax stats over N tokens of k.
// ===========================================================================
__global__ __launch_bounds__(256)
void k_softmax_stats_kernel(const unsigned short* __restrict__ qkv,
                            float* __restrict__ colmax, float* __restrict__ colsum) {
  const int bh = blockIdx.x;
  const int b = bh >> 3, hh = bh & 7;
  const int ch = threadIdx.x & 31;
  const int g  = threadIdx.x >> 5;
  const unsigned short* base = qkv + (size_t)b * N_ * QKV_LD + 256 + hh * 32 + ch;

  float m = -3.4e38f, s = 0.f;
  for (int n = g; n < N_; n += 8) {
    const float val = bfr2f(base[(size_t)n * QKV_LD]);
    if (val > m) { s = s * __expf(m - val); m = val; }
    s += __expf(val - m);
  }
  __shared__ float sm[8][32];
  __shared__ float ss[8][32];
  sm[g][ch] = m; ss[g][ch] = s;
  __syncthreads();
  if (g == 0) {
    float M = sm[0][ch], S = ss[0][ch];
#pragma unroll
    for (int i = 1; i < 8; ++i) {
      const float mi = sm[i][ch], si = ss[i][ch];
      const float Mn = fmaxf(M, mi);
      S = S * __expf(M - Mn) + si * __expf(mi - Mn);
      M = Mn;
    }
    colmax[b * 256 + hh * 32 + ch] = M;
    colsum[b * 256 + hh * 32 + ch] = S;
  }
}

// ===========================================================================
// Kernel 3: kv[b,h,kk,vv] = scale * (sum_n exp(k[n,kk]) * v[n,vv]) * exp(-max)/sumexp
// grid 64 (one block per (b,h)), 256 threads (8 waves).
// N processed in 48 chunks of 256 tokens: LDS-staged transposed bf16 tiles
// (exp folded in at staging; softmax normalization folded into the epilogue),
// then per-wave fragments are 8x ds_load_b128 + 4 WMMAs per chunk.
// ===========================================================================
__global__ __launch_bounds__(256)
void kv_gemm_kernel(const unsigned short* __restrict__ qkv,
                    const float* __restrict__ colmax, const float* __restrict__ colsum,
                    float* __restrict__ kv) {
  __shared__ __align__(16) union {
    struct { unsigned short kT[32][264]; unsigned short vT[32][264]; } t;  // 33.8 KB
    float red[8][4][256];                                                  // 32 KB
  } u;

  const int bh = blockIdx.x;
  const int b = bh >> 3, hh = bh & 7;
  const int t = threadIdx.x;
  const int lane = t & 31, wave = t >> 5;
  const int m = lane & 15, hi = lane >> 4;

  const unsigned short* kbase = qkv + (size_t)b * N_ * QKV_LD + 256 + hh * 32;
  const unsigned short* vbase = qkv + (size_t)b * N_ * QKV_LD + 512 + hh * 32;

  v8f acc00 = {}, acc01 = {}, acc10 = {}, acc11 = {};
  const int nb2 = wave * 32;                 // this wave's 32-token slice of a chunk

  for (int c = 0; c < N_ / 256; ++c) {
    const size_t row = (size_t)(c * 256 + t) * QKV_LD;
    const unsigned short* krow = kbase + row;
    const unsigned short* vrow = vbase + row;
#pragma unroll
    for (int g = 0; g < 4; ++g) {
      const v8bf k8 = ld8(krow + g * 8);
      const v8bf v8 = ld8(vrow + g * 8);
#pragma unroll
      for (int e = 0; e < 8; ++e) {
        u.t.kT[g * 8 + e][t] = f2bfraw(__expf(bfe2f(k8[e])));
        u.t.vT[g * 8 + e][t] = __builtin_bit_cast(unsigned short, v8[e]);
      }
    }
    __syncthreads();

    const unsigned short* a0p = &u.t.kT[m][nb2 + hi * 8];
    const unsigned short* a1p = &u.t.kT[16 + m][nb2 + hi * 8];
    const unsigned short* b0p = &u.t.vT[m][nb2 + hi * 8];
    const unsigned short* b1p = &u.t.vT[16 + m][nb2 + hi * 8];
    const v16bf a0 = frag_concat(ld8(a0p), ld8(a0p + 16));
    const v16bf a1 = frag_concat(ld8(a1p), ld8(a1p + 16));
    const v16bf b0 = frag_concat(ld8(b0p), ld8(b0p + 16));
    const v16bf b1 = frag_concat(ld8(b1p), ld8(b1p + 16));

    acc00 = WMMA_BF16(a0, b0, acc00);
    acc01 = WMMA_BF16(a0, b1, acc01);
    acc10 = WMMA_BF16(a1, b0, acc10);
    acc11 = WMMA_BF16(a1, b1, acc11);
    __syncthreads();
  }

  // cross-wave reduction (LDS reused; phases separated by the barrier below)
#pragma unroll
  for (int r = 0; r < 8; ++r) {
    u.red[wave][0][lane + 32 * r] = acc00[r];
    u.red[wave][1][lane + 32 * r] = acc01[r];
    u.red[wave][2][lane + 32 * r] = acc10[r];
    u.red[wave][3][lane + 32 * r] = acc11[r];
  }
  __syncthreads();

  const int sl = t & 31, sr = t >> 5;
  const int mrow = sr + ((sl >> 4) << 3);
  const int ncol = sl & 15;
  const float scale = 0.17677669529663687f;  // Ch^-0.5 = 1/sqrt(32)
#pragma unroll
  for (int tt = 0; tt < 4; ++tt) {
    float sum = 0.f;
#pragma unroll
    for (int w = 0; w < 8; ++w) sum += u.red[w][tt][t];
    const int kk = (tt >> 1) * 16 + mrow;
    const int vv = (tt & 1) * 16 + ncol;
    const int ci = b * 256 + hh * 32 + kk;
    // sum = Sum_n exp(k) * v ; normalize: / (colsum * exp(colmax)) , then * scale
    const float norm = scale * __expf(-colmax[ci]) / colsum[ci];
    kv[(((size_t)bh) * 32 + kk) * 32 + vv] = sum * norm;
  }
}

// ===========================================================================
// Kernel 4: fa[b*N+n, h*32+vv] = sum_kk q[n,kk] * kv[b,h,kk,vv]   (K=32: 1 WMMA)
// ===========================================================================
__global__ __launch_bounds__(256)
void factor_att_kernel(const unsigned short* __restrict__ qkv,
                       const float* __restrict__ kv,
                       unsigned short* __restrict__ fa) {
  __shared__ __align__(16) unsigned short kvT[8][32][32];  // [head][vv][kk], 16KB

  const int lane = threadIdx.x & 31, wave = threadIdx.x >> 5;
  const int gmBase = blockIdx.x * 128 + wave * 16;
  const int b = (blockIdx.x * 128) / N_;

  const float* kvb = kv + (size_t)b * 8 * 1024;
#pragma unroll
  for (int i = 0; i < 8; ++i) {
    const int idx = threadIdx.x + i * 256;     // 0..2047 float4s
    const int v4  = (idx & 7) * 4;
    const int kk  = (idx >> 3) & 31;
    const int hh  = idx >> 8;
    const float4 f = *(const float4*)(kvb + hh * 1024 + kk * 32 + v4);
    kvT[hh][v4 + 0][kk] = f2bfraw(f.x);
    kvT[hh][v4 + 1][kk] = f2bfraw(f.y);
    kvT[hh][v4 + 2][kk] = f2bfraw(f.z);
    kvT[hh][v4 + 3][kk] = f2bfraw(f.w);
  }
  __syncthreads();

  const int m = lane & 15, hi = lane >> 4, mhi = hi << 3;
  const unsigned short* qrow = qkv + (size_t)(gmBase + m) * QKV_LD;

  for (int hh = 0; hh < H_; ++hh) {
    const unsigned short* qk = qrow + hh * 32 + hi * 8;
    const v16bf a = frag_concat(ld8(qk), ld8(qk + 16));
    const unsigned short* k0p = &kvT[hh][m][hi * 8];
    const unsigned short* k1p = &kvT[hh][16 + m][hi * 8];
    const v16bf b0 = frag_concat(ld8(k0p), ld8(k0p + 16));
    const v16bf b1 = frag_concat(ld8(k1p), ld8(k1p + 16));
    v8f d0 = {}, d1 = {};
    d0 = WMMA_BF16(a, b0, d0);
    d1 = WMMA_BF16(a, b1, d1);
#pragma unroll
    for (int r = 0; r < 8; ++r) {
      const size_t rowOff = (size_t)(gmBase + mhi + r) * C_ + hh * 32 + m;
      fa[rowOff]      = f2bfraw(d0[r]);
      fa[rowOff + 16] = f2bfraw(d1[r]);
    }
  }
}

// ===========================================================================
// Kernel 5: out[b,c',n] = gamma * (fa @ W_proj + b_proj)[n,c'] + x[b,c',n]
// ===========================================================================
__global__ __launch_bounds__(256)
void proj_residual_kernel(const unsigned short* __restrict__ fa,
                          const unsigned short* __restrict__ WpT,
                          const float* __restrict__ bp,
                          const float* __restrict__ gamma_p, const float* __restrict__ x,
                          float* __restrict__ out) {
  const int lane = threadIdx.x & 31, wave = threadIdx.x >> 5;
  const int gmBase = blockIdx.x * 128 + wave * 16;
  const int col0 = blockIdx.y * 64;
  const int m = lane & 15, hi = lane >> 4;
  const unsigned short* farow = fa + (size_t)(gmBase + m) * C_;
  const unsigned short* wcol  = WpT + (size_t)(col0 + m) * C_;
  const float gamma = gamma_p[0];

  v8f acc0 = {}, acc1 = {}, acc2 = {}, acc3 = {};
  for (int k0 = 0; k0 < C_; k0 += 32) {
    const unsigned short* ak = farow + k0 + hi * 8;
    const v16bf a = frag_concat(ld8(ak), ld8(ak + 16));
    const unsigned short* wk = wcol + k0 + hi * 8;
    const v16bf b0 = frag_concat(ld8(wk),           ld8(wk + 16));
    const v16bf b1 = frag_concat(ld8(wk + 16 * C_), ld8(wk + 16 * C_ + 16));
    const v16bf b2 = frag_concat(ld8(wk + 32 * C_), ld8(wk + 32 * C_ + 16));
    const v16bf b3 = frag_concat(ld8(wk + 48 * C_), ld8(wk + 48 * C_ + 16));
    acc0 = WMMA_BF16(a, b0, acc0);
    acc1 = WMMA_BF16(a, b1, acc1);
    acc2 = WMMA_BF16(a, b2, acc2);
    acc3 = WMMA_BF16(a, b3, acc3);
  }

  const int mhi = hi << 3;
  const int b = gmBase / N_;
  const float bb0 = bp[col0 + m];
  const float bb1 = bp[col0 + 16 + m];
  const float bb2 = bp[col0 + 32 + m];
  const float bb3 = bp[col0 + 48 + m];
#pragma unroll
  for (int r = 0; r < 8; ++r) {
    const int gm = gmBase + mhi + r;
    const int n = gm - b * N_;
    const size_t base = (size_t)b * C_ * N_ + n;
    const size_t o0 = base + (size_t)(col0 + m) * N_;
    const size_t o1 = base + (size_t)(col0 + 16 + m) * N_;
    const size_t o2 = base + (size_t)(col0 + 32 + m) * N_;
    const size_t o3 = base + (size_t)(col0 + 48 + m) * N_;
    out[o0] = gamma * (acc0[r] + bb0) + x[o0];
    out[o1] = gamma * (acc1[r] + bb1) + x[o1];
    out[o2] = gamma * (acc2[r] + bb2) + x[o2];
    out[o3] = gamma * (acc3[r] + bb3) + x[o3];
  }
}

// ===========================================================================
extern "C" void kernel_launch(void* const* d_in, const int* in_sizes, int n_in,
                              void* d_out, int out_size, void* d_ws, size_t ws_size,
                              hipStream_t stream) {
  (void)in_sizes; (void)n_in; (void)out_size; (void)ws_size;
  const float* x     = (const float*)d_in[0];
  const float* Wqkv  = (const float*)d_in[1];
  const float* bqkv  = (const float*)d_in[2];
  const float* Wp    = (const float*)d_in[3];
  const float* bp    = (const float*)d_in[4];
  const float* gamma = (const float*)d_in[5];
  float* out = (float*)d_out;

  char* ws = (char*)d_ws;
  unsigned short* qkv = (unsigned short*)ws;                   // 151 MB
  size_t off = (size_t)B_ * N_ * QKV_LD * sizeof(unsigned short);
  float* colmax = (float*)(ws + off); off += 2048 * sizeof(float);
  float* colsum = (float*)(ws + off); off += 2048 * sizeof(float);
  float* kv     = (float*)(ws + off); off += (size_t)B_ * H_ * 32 * 32 * sizeof(float);
  unsigned short* fa    = (unsigned short*)(ws + off);         // 50 MB
  off += (size_t)B_ * N_ * C_ * sizeof(unsigned short);
  unsigned short* WqkvT = (unsigned short*)(ws + off);         // 768*256 bf16
  off += (size_t)QKV_LD * C_ * sizeof(unsigned short);
  unsigned short* WpT   = (unsigned short*)(ws + off);         // 256*256 bf16

  dim3 blk(256);
  transpose_bf16_kernel<<<dim3((QKV_LD * C_) / 256), blk, 0, stream>>>(Wqkv, WqkvT, C_, QKV_LD);
  transpose_bf16_kernel<<<dim3((C_ * C_) / 256),     blk, 0, stream>>>(Wp,   WpT,   C_, C_);
  qkv_gemm_kernel<<<dim3(768, 12), blk, 0, stream>>>(x, WqkvT, bqkv, qkv);
  k_softmax_stats_kernel<<<dim3(64), blk, 0, stream>>>(qkv, colmax, colsum);
  kv_gemm_kernel<<<dim3(64), blk, 0, stream>>>(qkv, colmax, colsum, kv);
  factor_att_kernel<<<dim3(768), blk, 0, stream>>>(qkv, kv, fa);
  proj_residual_kernel<<<dim3(768, 4), blk, 0, stream>>>(fa, WpT, bp, gamma, x, out);
}
